// RandomBinaryConv_87557203296393
// MI455X (gfx1250) — matrix operand order repaired and checked by reading, not compiled, not run on hardware
//
#include <hip/hip_runtime.h>

typedef __attribute__((ext_vector_type(16))) _Float16 v16h;
typedef __attribute__((ext_vector_type(8)))  float    v8f;

constexpr int IN_C  = 256;
constexpr int OUT_C = 256;
constexpr int Hh    = 56;
constexpr int Ww    = 56;
constexpr int NB    = 32;
constexpr int SP    = Hh * Ww;          // 3136 spatial per image
constexpr int NSP   = NB * SP;          // 100352 GEMM-N columns
constexpr int NCOLS = 128;              // columns per workgroup (8 N-tiles)
constexpr int NBLK  = NSP / NCOLS;      // 784 workgroups (exact)
constexpr int WSWZ  = 9 * 8 * 16 * 512; // swizzled weight halves (1.18 MB)

// ---------------------------------------------------------------------------
// Pre-pass: fp32 OIHW weights -> f16 in per-lane WMMA A-fragment order
// (16-bit A 16x32; this code's contiguous k order, not the ISA table order; B uses the same):
//   lanes 0-15 : M = lane,    K = 0..15
//   lanes 16-31: M = lane-16, K = 16..31
// Layout [tap][ic_chunk][oc_tile][lane][16 halves] -> 32 contiguous B/lane.
// ---------------------------------------------------------------------------
__global__ void swizzle_weights_kernel(const float* __restrict__ wgt,
                                       _Float16* __restrict__ wf) {
  int t = blockIdx.x * 256 + threadIdx.x;
  if (t >= WSWZ) return;
  int j      = t & 15;
  int lane   = (t >> 4) & 31;
  int octile = (t >> 9) & 15;
  int icc    = (t >> 13) & 7;
  int tap    = t >> 16;
  int m  = lane & 15;
  int kb = (lane < 16) ? 0 : 16;
  int K  = kb + j;
  int oc = octile * 16 + m;
  int ic = icc * 32 + K;
  wf[t] = (_Float16)wgt[(oc * IN_C + ic) * 9 + tap];   // OIHW, tap = kh*3+kw
}

// ---------------------------------------------------------------------------
// Implicit-GEMM conv3x3, V_WMMA_F32_16X16X32_F16, f32 accumulate.
// Workgroup: 512 thr = 16 waves; covers M=256 (all oc) x N=128 columns.
// Wave w: M-tiles {2*(w>>1), 2*(w>>1)+1} x N-tiles {4*(w&1) .. +3}
//   -> 8 accumulators, 8 WMMAs per (tap, ic-chunk).
// B staged in LDS in fragment order (lanes 0-15: K=0..15, lanes16-31: K=16..31):
//   Bfrag[ic_chunk:8][n_tile:8][lane:32][16 halves] = 64 KB.
// ---------------------------------------------------------------------------
__global__ __launch_bounds__(512)
void conv3x3_wmma_kernel(const float* __restrict__ x,
                         const _Float16* __restrict__ wf,
                         float* __restrict__ out) {
  __shared__ __align__(32) _Float16 Bfrag[8 * 8 * 512];  // 64 KB

  const int tid  = threadIdx.x;
  const int lane = tid & 31;
  const int wv   = tid >> 5;          // 0..15
  const int mg   = wv >> 1;           // 0..7 -> M-tiles 2mg, 2mg+1
  const int nh   = wv & 1;            // 0/1 -> N-tiles 4nh .. 4nh+3
  const int s0   = blockIdx.x * NCOLS;

  // ---- loader role: thread owns column (tid&127), K-row phase (tid>>7) ----
  const int lcol = tid & 127;
  const int krow = tid >> 7;          // 0..3; thread stages ic = krow + 4*j + 32*icc
  const int s  = s0 + lcol;
  const int n  = s / SP;
  const int hw = s - n * SP;
  const int h  = hw / Ww;
  const int w  = hw - h * Ww;
  // LDS slot base for (K, col): off = icc*4096 + nt*512 + ((K&16)<<4) + (col&15)*16 + (K&15)
  _Float16* bst = &Bfrag[((lcol >> 4) << 9) + ((lcol & 15) << 4) + krow];

  const v16h* Aw = (const v16h*)wf;

  v8f acc[2][4] = {};

  for (int tap = 0; tap < 9; ++tap) {
    const int dh = tap / 3 - 1;
    const int dw = tap % 3 - 1;
    const int y  = h + dh;
    const int xx = w + dw;
    const bool valid = ((unsigned)y < (unsigned)Hh) && ((unsigned)xx < (unsigned)Ww);
    const int yc = min(max(y, 0), Hh - 1);    // clamp: loads always in-bounds
    const int xc = min(max(xx, 0), Ww - 1);
    const float* src = x + (n * IN_C + krow) * SP + yc * Ww + xc;

    __syncthreads();                          // previous tap's B reads done
#pragma unroll
    for (int icc = 0; icc < 8; ++icc) {
#pragma unroll
      for (int j = 0; j < 8; ++j) {
        float v = src[(icc * 32 + 4 * j) * SP];     // ic = 32*icc + krow + 4*j
        v = valid ? v : 0.0f;                       // zero-pad halo
        // K = krow + 4*j: (K&16)<<4 == (j&4)<<6 ; K&15 == krow + ((j&3)<<2)
        bst[icc * 4096 + ((j & 4) << 6) + ((j & 3) << 2)] = (_Float16)v;
      }
    }
    __syncthreads();                          // B block visible to all waves

    const v16h* Ap = Aw + (tap * 128 + 2 * mg) * 32 + lane;       // A frags
    const v16h* Bp = (const v16h*)&Bfrag[(nh * 4) * 512 + lane * 16];
#pragma unroll
    for (int icc = 0; icc < 8; ++icc) {
      v16h a0 = Ap[icc * 512];
      v16h a1 = Ap[icc * 512 + 32];
#pragma unroll
      for (int j = 0; j < 4; ++j) {
        v16h b = Bp[icc * 256 + j * 32];
        acc[0][j] = __builtin_amdgcn_wmma_f32_16x16x32_f16(
            false, a0, false, b, (short)0, acc[0][j], false, false);
        acc[1][j] = __builtin_amdgcn_wmma_f32_16x16x32_f16(
            false, a1, false, b, (short)0, acc[1][j], false, false);
      }
    }
  }

  // ---- store C/D (f32 16x16: lanes 0-15 M=vgpr, lanes 16-31 M=8+vgpr) ----
  const int ccol  = lane & 15;
  const int mbase = (lane >> 4) << 3;
#pragma unroll
  for (int ntl = 0; ntl < 4; ++ntl) {
    const int cs  = s0 + (nh * 4 + ntl) * 16 + ccol;
    const int cn  = cs / SP;
    const int chw = cs - cn * SP;
#pragma unroll
    for (int m = 0; m < 2; ++m) {
      float* op = out + (cn * OUT_C + (2 * mg + m) * 16 + mbase) * SP + chw;
#pragma unroll
      for (int i = 0; i < 8; ++i) op[i * SP] = acc[m][ntl][i];
    }
  }
}

// ---------------------------------------------------------------------------
extern "C" void kernel_launch(void* const* d_in, const int* in_sizes, int n_in,
                              void* d_out, int out_size, void* d_ws, size_t ws_size,
                              hipStream_t stream) {
  const float* x   = (const float*)d_in[0];   // [32,256,56,56] fp32
  const float* wgt = (const float*)d_in[1];   // [256,256,3,3]  fp32 (+-1/0)
  float* out = (float*)d_out;                 // [32,256,56,56] fp32
  _Float16* wf = (_Float16*)d_ws;             // 1.18 MB swizzled f16 weights

  swizzle_weights_kernel<<<(WSWZ + 255) / 256, 256, 0, stream>>>(wgt, wf);
  conv3x3_wmma_kernel<<<NBLK, 512, 0, stream>>>(x, wf, out);
}
